// LocalizerHead_82291573391853
// MI455X (gfx1250) — matrix operand order repaired and checked by reading, not compiled, not run on hardware
//
#include <hip/hip_runtime.h>

// ---------------------------------------------------------------------------
// LocalizerHead for MI455X (gfx1250, wave32, WMMA, async LDS DMA).
//
//   K1: x_bf16[8192,512] = SiLU(BN(features[8192,1280] @ conv_kernel[1280,512]))
//       bf16 WMMA 16x16x32, fp32 accum. Output bf16 -> d_ws (8 MB, L2-resident).
//   K2: per (n, group of 4 p): logits[1024,40] = x_n @ W_np + b  (bf16 WMMA),
//       A streamed with GLOBAL_LOAD_ASYNC_TO_LDS_B128 into a double buffer,
//       logits kept in LDS (168 KB of 320 KB/WGP), fused softmax/decode.
// ---------------------------------------------------------------------------

typedef __attribute__((ext_vector_type(16))) __bf16 v16bf;
typedef __attribute__((ext_vector_type(8)))  float  v8f;

union Frag { v16bf v; uint4 q[2]; };

// ---- fp32 -> bf16 (prefer v_cvt_pk_bf16_f32 when the target has it) -------
#if __has_builtin(__builtin_amdgcn_cvt_pk_bf16_f32)
typedef __attribute__((ext_vector_type(2))) __bf16 v2bf;
__device__ __forceinline__ unsigned int pack2bf(float lo, float hi) {
  v2bf r = __builtin_amdgcn_cvt_pk_bf16_f32(lo, hi);
  return __builtin_bit_cast(unsigned int, r);
}
__device__ __forceinline__ unsigned short f2bf(float f) {
  v2bf r = __builtin_amdgcn_cvt_pk_bf16_f32(f, 0.0f);
  return (unsigned short)(__builtin_bit_cast(unsigned int, r) & 0xffffu);
}
#else
__device__ __forceinline__ unsigned short f2bf(float f) {
  unsigned a = __float_as_uint(f);
  a = a + 0x7fffu + ((a >> 16) & 1u);          // round-to-nearest-even
  return (unsigned short)(a >> 16);
}
__device__ __forceinline__ unsigned int pack2bf(float lo, float hi) {
  unsigned a = __float_as_uint(lo); a = a + 0x7fffu + ((a >> 16) & 1u);
  unsigned b = __float_as_uint(hi); b = b + 0x7fffu + ((b >> 16) & 1u);
  return (a >> 16) | (b & 0xffff0000u);
}
#endif

// ---- gfx1250 async global->LDS DMA (ASYNCcnt) -----------------------------
#if __has_builtin(__builtin_amdgcn_global_load_async_to_lds_b128)
#define HAVE_ASYNC 1
#else
#define HAVE_ASYNC 0
#endif

// builtin signature (from hipcc diagnostic): (int4 AS1*, int4 AS3*, Imm, Imm)
typedef int v4i_vec __attribute__((vector_size(16)));
typedef __attribute__((address_space(1))) v4i_vec gv4i;
typedef __attribute__((address_space(3))) v4i_vec lv4i;
__device__ __forceinline__ gv4i* to_gbl4(const void* p) {
  return (gv4i*)(unsigned long long)p;
}
__device__ __forceinline__ lv4i* to_lds4(const void* p) {
  // ISA: generic->LDS aperture address truncates to low 32 bits
  return (lv4i*)(unsigned int)(unsigned long long)p;
}

#if __has_builtin(__builtin_amdgcn_s_wait_asynccnt)
#define WAIT_ASYNC(n) __builtin_amdgcn_s_wait_asynccnt(n)
#else
#define WAIT_ASYNC(n) asm volatile("s_wait_asynccnt %0" :: "i"(n) : "memory")
#endif

// ========================= Kernel 1: conv1x1 + BN + SiLU ====================
// A tile in LDS: 128 rows x 32 k, bf16, row stride 80 B (16B aligned)
//   -> fragment = 2x ds_load_b128: lanes<16 read k-chunks {0..7,16..23},
//      lanes>=16 read {8..15,24..31}  (ISA 16-bit A 16x32 layout).
// B tile in LDS: 128 cols x 32 k, bf16 column-major, col stride 80 B
//   -> fragment = 2x ds_load_b128: lanes<16 k={0..7,16..23}, lanes>=16 k={8..15,24..31} (same as A).
__global__ __launch_bounds__(256) void k1_conv_bn_silu(
    const float* __restrict__ feat, const float* __restrict__ ck,
    const float* __restrict__ gamma, const float* __restrict__ beta,
    const float* __restrict__ mean,  const float* __restrict__ var,
    unsigned short* __restrict__ xout)
{
  __shared__ unsigned char sA[128 * 80];   // 10.0 KB
  __shared__ unsigned char sB[128 * 80];   // 10.0 KB
  const int m0 = blockIdx.x * 128;
  const int n0 = blockIdx.y * 128;
  const int t = threadIdx.x;
  const int wave = t >> 5, lane = t & 31;
  const int wm = (wave >> 2) * 64;   // 2 waves in M (64 rows each)
  const int wn = (wave & 3) * 32;    // 4 waves in N (32 cols each)
  const int l15 = lane & 15;
  const int hi  = lane >> 4;

  const v8f vz = {0.f,0.f,0.f,0.f,0.f,0.f,0.f,0.f};
  v8f acc[4][2];
#pragma unroll
  for (int i = 0; i < 4; i++) { acc[i][0] = vz; acc[i][1] = vz; }

  for (int k0 = 0; k0 < 1280; k0 += 32) {
    { // stage A: 128x32 fp32 -> bf16, row-major 64B rows, ds_store_b64 x4
      const int row = t >> 1, half = t & 1;
      const float* src = feat + (size_t)(m0 + row) * 1280 + k0 + half * 16;
      uint2* dst = (uint2*)(sA + row * 80 + half * 32);
#pragma unroll
      for (int j = 0; j < 4; j++) {
        float4 f = *(const float4*)(src + j * 4);
        dst[j] = make_uint2(pack2bf(f.x, f.y), pack2bf(f.z, f.w));
      }
    }
    { // stage B: 32x128 fp32 (row-major k x d) -> bf16 column-major
      const int kr = t >> 3, cs = (t & 7) * 16;
      const float* src = ck + (size_t)(k0 + kr) * 512 + n0 + cs;
#pragma unroll
      for (int j = 0; j < 4; j++) {
        float4 f = *(const float4*)(src + j * 4);
        *(unsigned short*)(sB + (cs + j*4 + 0) * 80 + kr * 2) = f2bf(f.x);
        *(unsigned short*)(sB + (cs + j*4 + 1) * 80 + kr * 2) = f2bf(f.y);
        *(unsigned short*)(sB + (cs + j*4 + 2) * 80 + kr * 2) = f2bf(f.z);
        *(unsigned short*)(sB + (cs + j*4 + 3) * 80 + kr * 2) = f2bf(f.w);
      }
    }
    __syncthreads();

    Frag b[2];
#pragma unroll
    for (int ni = 0; ni < 2; ni++) {
      const unsigned char* p = sB + (wn + ni * 16 + l15) * 80 + hi * 16;
      b[ni].q[0] = *(const uint4*)(p);
      b[ni].q[1] = *(const uint4*)(p + 32);
    }
#pragma unroll
    for (int mi = 0; mi < 4; mi++) {
      Frag a;
      const unsigned char* p = sA + (wm + mi * 16 + l15) * 80 + hi * 16;
      a.q[0] = *(const uint4*)(p);
      a.q[1] = *(const uint4*)(p + 32);
#pragma unroll
      for (int ni = 0; ni < 2; ni++)
        acc[mi][ni] = __builtin_amdgcn_wmma_f32_16x16x32_bf16(
            false, a.v, false, b[ni].v, (short)0, acc[mi][ni], false, false);
    }
    __syncthreads();
  }

  // epilogue: BN + SiLU, store bf16
#pragma unroll
  for (int ni = 0; ni < 2; ni++) {
    const int gn = n0 + wn + ni * 16 + l15;
    const float mu = mean[gn];
    const float sc = gamma[gn] * rsqrtf(var[gn] + 1e-3f);
    const float bt = beta[gn];
#pragma unroll
    for (int mi = 0; mi < 4; mi++) {
#pragma unroll
      for (int r = 0; r < 8; r++) {
        const int gm = m0 + wm + mi * 16 + r + hi * 8;
        float y = (acc[mi][ni][r] - mu) * sc + bt;
        y = y * (1.0f / (1.0f + __expf(-y)));          // SiLU
        xout[(size_t)gm * 512 + gn] = f2bf(y);
      }
    }
  }
}

// ================= Kernel 2: per-(n, 4p) GEMM + softmax decode ==============
// Dynamic LDS layout (214,272 B total):
//   [0      , 81920 )  A stage buffer 0: 1024 rows x 32 k bf16, stride 80 B
//   [81920  , 163840)  A stage buffer 1 (double buffer for async DMA)
//   [163840 , 213760)  B: 48 cols x 512 k bf16 col-major, stride 1040 B
//   [0      , 172032)  logits (aliases A0/A1/B head after GEMM): 1024 x 42 fp32
//   [213760 , 214272)  reduction scratch (64 floats used)
#define K2_A_BYTES 81920
#define K2_B_OFF   163840
#define K2_BSTRIDE 1040
#define K2_LSTRIDE 168
#define K2_RED_OFF 213760
#define K2_DYN     214272

__global__ __launch_bounds__(256) void k2_head(
    const unsigned short* __restrict__ x, const float* __restrict__ wts,
    float* __restrict__ out)
{
  extern __shared__ unsigned char dsm[];
  const int n  = blockIdx.y;
  const int p0 = blockIdx.x * 4;
  const int t = threadIdx.x;
  const int wave = t >> 5, lane = t & 31;
  const int l15 = lane & 15, hi = lane >> 4;
  const size_t wbase = (size_t)n * (512 * 5130) + (size_t)p0 * 5130;
  const size_t xbase = (size_t)n * 1024 * 512;

  // issue/copy one 1024x32 bf16 A chunk into buffer `buf` (16 ops per wave)
  auto issueA = [&](int kt, int buf) {
    unsigned char* dst = dsm + buf * K2_A_BYTES;
    const unsigned short* src = x + xbase + kt;
    for (int cid = t; cid < 4096; cid += 256) {
      const int row = cid >> 2, ch = cid & 3;
#if HAVE_ASYNC
      __builtin_amdgcn_global_load_async_to_lds_b128(
          to_gbl4(src + (size_t)row * 512 + ch * 8),
          to_lds4(dst + row * 80 + ch * 16), 0, 0);
#else
      *(uint4*)(dst + row * 80 + ch * 16) =
          *(const uint4*)(src + (size_t)row * 512 + ch * 8);
#endif
    }
  };

  // kick off DMA for the first A chunk; it overlaps the strided B gather below
  issueA(0, 0);

  // zero B region (cols 40..47 are pad; keep them clean)
  for (int i = t; i < (48 * K2_BSTRIDE) / 4; i += 256)
    ((unsigned int*)(dsm + K2_B_OFF))[i] = 0u;
  __syncthreads();
  // gather B: 512 x 40 fp32 (stride-10 channel layout) -> bf16 col-major
  for (int e = t; e < 512 * 40; e += 256) {
    const int col = e >> 9;
    const int c   = e & 511;
    const int pq = col / 10, C = col - pq * 10;
    const float f = wts[wbase + (size_t)pq * 5130 + c * 10 + C];
    *(unsigned short*)(dsm + K2_B_OFF + col * K2_BSTRIDE + c * 2) = f2bf(f);
  }

  const v8f vz = {0.f,0.f,0.f,0.f,0.f,0.f,0.f,0.f};
  v8f acc[8][3];
#pragma unroll
  for (int i = 0; i < 8; i++) { acc[i][0] = vz; acc[i][1] = vz; acc[i][2] = vz; }

  for (int kt = 0; kt < 512; kt += 32) {
    const int buf = (kt >> 5) & 1;
    if (kt + 32 < 512) {
      issueA(kt + 32, buf ^ 1);          // DMA next chunk while computing this one
      WAIT_ASYNC(16);                    // current chunk's 16 loads complete (in-order)
    } else {
      WAIT_ASYNC(0);
    }
    __syncthreads();

    const unsigned char* abuf = dsm + buf * K2_A_BYTES;
    Frag b[3];
#pragma unroll
    for (int ni = 0; ni < 3; ni++) {
      const unsigned char* p =
          dsm + K2_B_OFF + (ni * 16 + l15) * K2_BSTRIDE + (kt + hi * 8) * 2;
      b[ni].q[0] = *(const uint4*)(p);
      b[ni].q[1] = *(const uint4*)(p + 32);
    }
#pragma unroll
    for (int mi = 0; mi < 8; mi++) {
      Frag a;
      const unsigned char* p = abuf + (wave * 128 + mi * 16 + l15) * 80 + hi * 16;
      a.q[0] = *(const uint4*)(p);
      a.q[1] = *(const uint4*)(p + 32);
#pragma unroll
      for (int ni = 0; ni < 3; ni++)
        acc[mi][ni] = __builtin_amdgcn_wmma_f32_16x16x32_bf16(
            false, a.v, false, b[ni].v, (short)0, acc[mi][ni], false, false);
    }
    __syncthreads();
  }

  // epilogue: logits + bias -> LDS (aliases dead A/B staging)
#pragma unroll
  for (int ni = 0; ni < 3; ni++) {
    const int col = ni * 16 + l15;
    if (col < 40) {
      const int pq = col / 10, C = col - pq * 10;
      const float bias = wts[wbase + (size_t)pq * 5130 + 5120 + C];
#pragma unroll
      for (int mi = 0; mi < 8; mi++) {
#pragma unroll
        for (int r = 0; r < 8; r++) {
          const int row = wave * 128 + mi * 16 + r + hi * 8;
          *(float*)(dsm + row * K2_LSTRIDE + col * 4) = acc[mi][ni][r] + bias;
        }
      }
    }
  }
  __syncthreads();

  // fused softmax decode per local p (q = 0..3); rows: hw = h*32 + w
  float* sb = (float*)(dsm + K2_RED_OFF);
  for (int q = 0; q < 4; q++) {
    // pass 1: maxes (exact 2-pass softmax)
    float m1 = -3.4e38f, mz = -3.4e38f;
    for (int i = 0; i < 4; i++) {
      const int row = t + i * 256;
      const float* L = (const float*)(dsm + row * K2_LSTRIDE + q * 40);
      m1 = fmaxf(m1, L[1]);
#pragma unroll
      for (int z = 0; z < 8; z++) mz = fmaxf(mz, L[2 + z]);
    }
#pragma unroll
    for (int s = 16; s > 0; s >>= 1) {
      m1 = fmaxf(m1, __shfl_xor(m1, s, 32));
      mz = fmaxf(mz, __shfl_xor(mz, s, 32));
    }
    if (lane == 0) { sb[wave] = m1; sb[8 + wave] = mz; }
    __syncthreads();
    float M1 = -3.4e38f, MZ = -3.4e38f;
#pragma unroll
    for (int w = 0; w < 8; w++) { M1 = fmaxf(M1, sb[w]); MZ = fmaxf(MZ, sb[8 + w]); }
    __syncthreads();

    // pass 2: exp-sums and expectation numerators
    float Z1 = 0, Sw1 = 0, Sh1 = 0, Zz = 0, Sw = 0, Sh = 0, Sz = 0, Su = 0;
    for (int i = 0; i < 4; i++) {
      const int row = t + i * 256;
      const float rw = (float)(row & 31) * (1.0f / 31.0f);
      const float rh = (float)(row >> 5) * (1.0f / 31.0f);
      const float* L = (const float*)(dsm + row * K2_LSTRIDE + q * 40);
      const float e1 = __expf(L[1] - M1);
      Z1 += e1; Sw1 += e1 * rw; Sh1 += e1 * rh;
      float rz = 0, rsz = 0;
#pragma unroll
      for (int z = 0; z < 8; z++) {
        const float e = __expf(L[2 + z] - MZ);
        rz += e; rsz += e * ((float)z * (1.0f / 7.0f));
      }
      Zz += rz; Sw += rz * rw; Sh += rz * rh; Sz += rsz; Su += L[0] * rz;
    }
    float v[8] = {Z1, Sw1, Sh1, Zz, Sw, Sh, Sz, Su};
#pragma unroll
    for (int j = 0; j < 8; j++)
#pragma unroll
      for (int s = 16; s > 0; s >>= 1) v[j] += __shfl_xor(v[j], s, 32);
    if (lane == 0) {
#pragma unroll
      for (int j = 0; j < 8; j++) sb[wave * 8 + j] = v[j];
    }
    __syncthreads();
    if (t == 0) {
      float a[8] = {0, 0, 0, 0, 0, 0, 0, 0};
      for (int w = 0; w < 8; w++)
        for (int j = 0; j < 8; j++) a[j] += sb[w * 8 + j];
      const float cx1 = a[1] / a[0], cy1 = a[2] / a[0];         // sm_xy decode (w,h)
      const float x2 = a[4] / a[3], y2 = a[5] / a[3], z2 = a[6] / a[3];
      const float U  = a[7] / a[3];
      const float sp = (U > 20.0f) ? U : log1pf(__expf(U));     // softplus
      const int p = p0 + q;
      const int idx = n * 512 + p;
      out[idx * 2 + 0] = x2 * 248.0f + 4.0f;                    // coords2d_pred
      out[idx * 2 + 1] = y2 * 248.0f + 4.0f;
      float* o3 = out + 8192;                                   // coords3d_rel_pred
      o3[idx * 3 + 0] = (cx1 * 248.0f + 4.0f) * (2.2f / 256.0f);
      o3[idx * 3 + 1] = (cy1 * 248.0f + 4.0f) * (2.2f / 256.0f);
      o3[idx * 3 + 2] = z2 * 2.2f;
      out[8192 + 12288 + idx] = sp + 0.01f;                     // uncertainties
    }
    __syncthreads();
  }
}

// =========================== host launch ====================================
extern "C" void kernel_launch(void* const* d_in, const int* in_sizes, int n_in,
                              void* d_out, int out_size, void* d_ws, size_t ws_size,
                              hipStream_t stream) {
  const float* feat = (const float*)d_in[0];   // (8,32,32,1280)
  const float* wts  = (const float*)d_in[1];   // (8,512,5130)
  const float* ck   = (const float*)d_in[2];   // (1,1,1280,512)
  const float* g    = (const float*)d_in[3];
  const float* b    = (const float*)d_in[4];
  const float* mu   = (const float*)d_in[5];
  const float* var  = (const float*)d_in[6];
  unsigned short* xbf = (unsigned short*)d_ws; // 8192x512 bf16 (8 MB)

  dim3 g1(64, 4);   // 8192/128 M-tiles x 512/128 N-tiles
  k1_conv_bn_silu<<<g1, 256, 0, stream>>>(feat, ck, g, b, mu, var, xbf);

  dim3 g2(128, 8);  // 128 p-groups x 8 n
  k2_head<<<g2, 256, K2_DYN, stream>>>(xbf, wts, (float*)d_out);
}
